// RotaryAttentionBlock_2680059593192
// MI455X (gfx1250) — matrix operand order repaired and checked by reading, not compiled, not run on hardware
//
#include <hip/hip_runtime.h>

// ---------------- types ----------------
typedef __bf16 bf16;
typedef __attribute__((ext_vector_type(16))) bf16  v16bf;
typedef __attribute__((ext_vector_type(8)))  bf16  v8bf;
typedef __attribute__((ext_vector_type(8)))  float v8f;

// pointer types for the async global->LDS builtin
typedef int v4i __attribute__((vector_size(16)));
typedef __attribute__((address_space(1))) v4i as1_v4i;  // global
typedef __attribute__((address_space(3))) v4i as3_v4i;  // LDS

#define WMMA_BF16(a, b, c) \
  __builtin_amdgcn_wmma_f32_16x16x32_bf16(false, (a), false, (b), (short)0, (c), false, false)

// Problem constants
static constexpr int Bz  = 2;
static constexpr int Tt  = 2048;
static constexpr int HID = 2048;
static constexpr int NH  = 32;
static constexpr int NKV = 8;
static constexpr int HD  = 64;
static constexpr int Mrows = Bz * Tt;  // 4096

// ---------------- fragment loads ----------------
static __device__ __forceinline__ v16bf ld_frag(const bf16* lo, const bf16* hi) {
  v8bf a = *(const v8bf*)lo;
  v8bf b = *(const v8bf*)hi;
  v16bf r;
#pragma unroll
  for (int i = 0; i < 8; ++i) { r[i] = a[i]; r[i + 8] = b[i]; }
  return r;
}
static __device__ __forceinline__ v16bf ld_b(const bf16* p) { return ld_frag(p, p + 8); }

// Row-max over the 32 score columns: lane-local max of the two tiles, then a
// 4-step XOR butterfly across the 16-lane column group. Uses v_permlane16_b32
// (pure VALU, no DS round-trips); step-major across all 8 rows.
static __device__ __forceinline__ v8f rowmax32(v8f s0, v8f s1) {
  v8f part;
#pragma unroll
  for (int i = 0; i < 8; ++i) part[i] = fmaxf(s0[i], s1[i]);
#if __has_builtin(__builtin_amdgcn_permlane16)
  const unsigned selA[4] = {0x67452301u, 0x54761032u, 0x32107654u, 0xFEDCBA98u};  // lanes 0-7
  const unsigned selB[4] = {0xEFCDAB89u, 0xDCFE98BAu, 0xBA98FEDCu, 0x76543210u};  // lanes 8-15
#pragma unroll
  for (int st = 0; st < 4; ++st) {
    v8f other;
#pragma unroll
    for (int i = 0; i < 8; ++i) {
      int v = __float_as_int(part[i]);
      int o = __builtin_amdgcn_permlane16(v, v, selA[st], selB[st], false, false);
      other[i] = __int_as_float(o);
    }
#pragma unroll
    for (int i = 0; i < 8; ++i) part[i] = fmaxf(part[i], other[i]);
  }
#else
#pragma unroll
  for (int m = 1; m < 16; m <<= 1) {
    v8f other;
#pragma unroll
    for (int i = 0; i < 8; ++i) other[i] = __shfl_xor(part[i], m, 32);
#pragma unroll
    for (int i = 0; i < 8; ++i) part[i] = fmaxf(part[i], other[i]);
  }
#endif
  return part;
}

// ---------------- CDNA5 async global->LDS copy (ASYNCcnt path) ----------------
static __device__ __forceinline__ void async_copy_b128(const bf16* g, const bf16* l) {
#if __has_builtin(__builtin_amdgcn_global_load_async_to_lds_b128)
  as1_v4i* gp = (as1_v4i*)(unsigned long long)(const void*)g;
  as3_v4i* lp = (as3_v4i*)(unsigned)(unsigned long long)(const void*)l;  // low 32 bits == LDS offset
  __builtin_amdgcn_global_load_async_to_lds_b128(gp, lp, 0, 0);
#else
  unsigned lo = (unsigned)(unsigned long long)l;
  unsigned long long ga = (unsigned long long)g;
  asm volatile("global_load_async_to_lds_b128 %0, %1, off" :: "v"(lo), "v"(ga) : "memory");
#endif
}
static __device__ __forceinline__ void wait_async(int pending) {
#if __has_builtin(__builtin_amdgcn_s_wait_asynccnt)
  if (pending) __builtin_amdgcn_s_wait_asynccnt(1);
  else         __builtin_amdgcn_s_wait_asynccnt(0);
#else
  if (pending) asm volatile("s_wait_asynccnt 1" ::: "memory");
  else         asm volatile("s_wait_asynccnt 0" ::: "memory");
#endif
}

// ---------------- f32 -> bf16 ----------------
__global__ void cvt_f32_bf16(const float* __restrict__ in, bf16* __restrict__ out, size_t n) {
  size_t i = (size_t)blockIdx.x * blockDim.x + threadIdx.x;
  size_t stride = (size_t)gridDim.x * blockDim.x;
  for (; i < n; i += stride) out[i] = (bf16)in[i];
}

// ---------------- fused WMMA GEMM: C = X[M,K] @ W[N,K]^T with epilogues -------------
// MODE 0: RoPE + scale, bf16 out [B, nh, T, HD]
// MODE 2: transpose,    bf16 out [B, nh, HD, T]
// MODE 3: plain f32 out [M, N]
// grid: (N/64, M/256); block 256 threads = 8 waves; wave tile 32x64.
// Shared 64x32 B tile double-buffered in LDS via async global->LDS copies.
template <int MODE>
__global__ void gemm_fused(const bf16* __restrict__ X, const bf16* __restrict__ W,
                           const int* __restrict__ pos, void* __restrict__ OUT,
                           int M, int N, int K, int nh, float scale) {
  __shared__ __align__(16) bf16 Bsm[2][64 * 32];

  const int tid  = threadIdx.x;
  const int lane = tid & 31;
  const int wave = tid >> 5;
  const int col  = lane & 15;
  const int half = lane >> 4;
  const int m0 = blockIdx.y * 256 + wave * 32;
  const int n0 = blockIdx.x * 64;

  // async staging assignment: 256 threads x 16B = one 64x32 bf16 tile
  const int r = tid >> 2;       // 0..63  (B row = weight row n0+r)
  const int c = tid & 3;        // 0..3   (16B chunk within the 32-wide K slice)
  const bf16* gsrc = W + (size_t)(n0 + r) * K + c * 8;
  bf16* ldst0 = &Bsm[0][r * 32 + c * 8];
  bf16* ldst1 = &Bsm[1][r * 32 + c * 8];

  v8f zero = {0.f, 0.f, 0.f, 0.f, 0.f, 0.f, 0.f, 0.f};
  v8f acc[2][4];
#pragma unroll
  for (int rt = 0; rt < 2; ++rt)
#pragma unroll
    for (int t = 0; t < 4; ++t) acc[rt][t] = zero;

  const bf16* xrow0 = X + (size_t)(m0 + col) * K;
  const bf16* xrow1 = X + (size_t)(m0 + 16 + col) * K;

  async_copy_b128(gsrc, ldst0);           // prime k0 = 0 into buf 0
  int buf = 0;
  for (int k0 = 0; k0 < K; k0 += 32) {
    const int more = (k0 + 32) < K;
    if (more) async_copy_b128(gsrc + k0 + 32, buf ? ldst0 : ldst1);
    wait_async(more);                     // current buffer's copy complete (in-order)
    __syncthreads();                      // ... for every wave in the block

    v16bf a0 = ld_frag(xrow0 + k0 + half * 16, xrow0 + k0 + half * 16 + 8);
    v16bf a1 = ld_frag(xrow1 + k0 + half * 16, xrow1 + k0 + half * 16 + 8);
    const bf16* bbase = &Bsm[buf][0];
#pragma unroll
    for (int t = 0; t < 4; ++t) {
      v16bf b = ld_b(bbase + (t * 16 + col) * 32 + half * 16);
      acc[0][t] = WMMA_BF16(a0, b, acc[0][t]);
      acc[1][t] = WMMA_BF16(a1, b, acc[1][t]);
    }
    __syncthreads();                      // all reads done before buffer is re-filled
    buf ^= 1;
  }

  // ---------------- epilogues ----------------
  if (MODE == 3) {
    float* C = (float*)OUT;
#pragma unroll
    for (int rt = 0; rt < 2; ++rt)
#pragma unroll
      for (int t = 0; t < 4; ++t)
#pragma unroll
        for (int i = 0; i < 8; ++i)
          C[(size_t)(m0 + rt * 16 + i + 8 * half) * N + n0 + t * 16 + col] = acc[rt][t][i];
  } else {
    bf16* O = (bf16*)OUT;
    const int head = n0 >> 6;  // 64-col tile == one head
#pragma unroll
    for (int rt = 0; rt < 2; ++rt)
#pragma unroll
      for (int i = 0; i < 8; ++i) {
        const int m   = m0 + rt * 16 + i + 8 * half;
        const int b   = m >> 11;            // / Tt
        const int tok = m & (Tt - 1);
        if (MODE == 2) {
#pragma unroll
          for (int t = 0; t < 4; ++t)
            O[(((size_t)b * nh + head) * HD + t * 16 + col) * Tt + tok] = (bf16)acc[rt][t][i];
        } else {  // MODE 0: RoPE (+scale)
          const float p = (float)pos[m];
#pragma unroll
          for (int t = 0; t < 4; ++t) {
            const int d = t * 16 + col;
            const float freq = __powf(10000.0f, -(float)(2 * (d >> 1)) * (1.0f / HD));
            const float ang = p * freq;
            const float rot = (t < 2) ? -acc[rt][t + 2][i] : acc[rt][t - 2][i];
            const float val = (acc[rt][t][i] * __cosf(ang) + rot * __sinf(ang)) * scale;
            O[(((size_t)b * nh + head) * Tt + tok) * HD + d] = (bf16)val;
          }
        }
      }
  }
}

// ---------------- flash attention: one wave per 16-row Q tile ----------------
// Qb [B,NH,T,HD] (pre-scaled), Kb [B,NKV,T,HD], Vt [B,NKV,HD,T], Ob [B,T,NH*HD] bf16
// Row sums via WMMA against an all-ones matrix; row max via permlane16 butterfly.
__global__ void attn_wmma(const bf16* __restrict__ Qb, const bf16* __restrict__ Kb,
                          const bf16* __restrict__ Vt, bf16* __restrict__ Ob) {
  __shared__ __align__(16) bf16 Pl[16 * 32];

  const int lane = threadIdx.x;
  const int col  = lane & 15;
  const int half = lane >> 4;
  const int q0 = blockIdx.x * 16;
  const int h  = blockIdx.y;
  const int b  = blockIdx.z;
  const int hk = h / (NH / NKV);

  const bf16* Q = Qb + (((size_t)b * NH + h) * Tt + q0) * HD;
  const bf16* K = Kb + ((size_t)b * NKV + hk) * Tt * HD;
  const bf16* V = Vt + ((size_t)b * NKV + hk) * HD * Tt;

  const bf16* qrow = Q + (size_t)col * HD;
  v16bf qa0 = ld_frag(qrow + half * 16,      qrow + half * 16 + 8);
  v16bf qa1 = ld_frag(qrow + 32 + half * 16, qrow + 32 + half * 16 + 8);

  v16bf ones;
#pragma unroll
  for (int i = 0; i < 16; ++i) ones[i] = (bf16)1.0f;

  v8f zero = {0.f, 0.f, 0.f, 0.f, 0.f, 0.f, 0.f, 0.f};
  v8f o[4];
#pragma unroll
  for (int t = 0; t < 4; ++t) o[t] = zero;
  v8f mrow, lrow;
#pragma unroll
  for (int i = 0; i < 8; ++i) { mrow[i] = -3.0e38f; lrow[i] = 0.0f; }

  for (int kb = 0; kb < Tt; kb += 32) {
    // S = Q @ K^T, 16x32 tile (two 16-col subtiles)
    v8f s0 = zero, s1 = zero;
    {
      const bf16* kr0 = K + (size_t)(kb + col) * HD + half * 16;
      const bf16* kr1 = K + (size_t)(kb + 16 + col) * HD + half * 16;
      s0 = WMMA_BF16(qa0, ld_b(kr0), s0);
      s0 = WMMA_BF16(qa1, ld_b(kr0 + 32), s0);
      s1 = WMMA_BF16(qa0, ld_b(kr1), s1);
      s1 = WMMA_BF16(qa1, ld_b(kr1 + 32), s1);
    }

    // online softmax state update (all elementwise in C-layout; max via permlane)
    v8f mx = rowmax32(s0, s1);
    v8f mnew, alpha;
#pragma unroll
    for (int i = 0; i < 8; ++i) {
      mnew[i]  = fmaxf(mrow[i], mx[i]);
      alpha[i] = __expf(mrow[i] - mnew[i]);
    }
    mrow = mnew;

    __syncthreads();  // WAR: previous iteration's P reads complete
#pragma unroll
    for (int i = 0; i < 8; ++i) {
      float p0 = __expf(s0[i] - mnew[i]);
      float p1 = __expf(s1[i] - mnew[i]);
      Pl[(i + 8 * half) * 32 + col]      = (bf16)p0;
      Pl[(i + 8 * half) * 32 + 16 + col] = (bf16)p1;
    }
#pragma unroll
    for (int t = 0; t < 4; ++t)
#pragma unroll
      for (int i = 0; i < 8; ++i) o[t][i] *= alpha[i];
    __syncthreads();  // publish P

    // P as A-fragment; row sums via WMMA with ones; PV accumulate
    const bf16* prow = Pl + col * 32;
    v16bf pa = ld_frag(prow + half * 16, prow + half * 16 + 8);

    v8f sums = WMMA_BF16(pa, ones, zero);   // D[r][*] = rowsum(P[r][:]) replicated
#pragma unroll
    for (int i = 0; i < 8; ++i) lrow[i] = lrow[i] * alpha[i] + sums[i];

#pragma unroll
    for (int t = 0; t < 4; ++t) {
      v16bf vf = ld_b(V + (size_t)(t * 16 + col) * Tt + kb + half * 16);
      o[t] = WMMA_BF16(pa, vf, o[t]);
    }
  }

#pragma unroll
  for (int t = 0; t < 4; ++t)
#pragma unroll
    for (int i = 0; i < 8; ++i) {
      float val = o[t][i] / lrow[i];
      Ob[((size_t)(b * Tt + q0 + i + 8 * half)) * (NH * HD) + h * HD + t * 16 + col] = (bf16)val;
    }
}

// ---------------- workspace layout (bytes) ----------------
static constexpr size_t OFF_XB  = 0;                                       // x bf16
static constexpr size_t OFF_WQB = OFF_XB  + (size_t)Mrows * HID * 2;
static constexpr size_t OFF_WKB = OFF_WQB + (size_t)(NH * HD) * HID * 2;
static constexpr size_t OFF_WVB = OFF_WKB + (size_t)(NKV * HD) * HID * 2;
static constexpr size_t OFF_WOB = OFF_WVB + (size_t)(NKV * HD) * HID * 2;
static constexpr size_t OFF_QB  = OFF_WOB + (size_t)HID * (NH * HD) * 2;   // roped Q bf16
static constexpr size_t OFF_KB  = OFF_QB  + (size_t)Mrows * (NH * HD) * 2; // roped K bf16
static constexpr size_t OFF_VT  = OFF_KB  + (size_t)Mrows * (NKV * HD) * 2;// V^T bf16
static constexpr size_t OFF_AB  = OFF_VT  + (size_t)Mrows * (NKV * HD) * 2;// attn out bf16

extern "C" void kernel_launch(void* const* d_in, const int* in_sizes, int n_in,
                              void* d_out, int out_size, void* d_ws, size_t ws_size,
                              hipStream_t stream) {
  (void)in_sizes; (void)n_in; (void)out_size; (void)ws_size;
  const float* x   = (const float*)d_in[0];
  const int*   pos = (const int*)d_in[1];
  const float* Wq  = (const float*)d_in[2];
  const float* Wk  = (const float*)d_in[3];
  const float* Wv  = (const float*)d_in[4];
  const float* Wo  = (const float*)d_in[5];
  float* out = (float*)d_out;
  char*  ws  = (char*)d_ws;

  bf16* xb  = (bf16*)(ws + OFF_XB);
  bf16* wqb = (bf16*)(ws + OFF_WQB);
  bf16* wkb = (bf16*)(ws + OFF_WKB);
  bf16* wvb = (bf16*)(ws + OFF_WVB);
  bf16* wob = (bf16*)(ws + OFF_WOB);
  bf16* qb  = (bf16*)(ws + OFF_QB);
  bf16* kb  = (bf16*)(ws + OFF_KB);
  bf16* vt  = (bf16*)(ws + OFF_VT);
  bf16* ab  = (bf16*)(ws + OFF_AB);

  // 1) bf16 conversions
  cvt_f32_bf16<<<2048, 256, 0, stream>>>(x,  xb,  (size_t)Mrows * HID);
  cvt_f32_bf16<<<2048, 256, 0, stream>>>(Wq, wqb, (size_t)(NH * HD) * HID);
  cvt_f32_bf16<<<1024, 256, 0, stream>>>(Wk, wkb, (size_t)(NKV * HD) * HID);
  cvt_f32_bf16<<<1024, 256, 0, stream>>>(Wv, wvb, (size_t)(NKV * HD) * HID);
  cvt_f32_bf16<<<2048, 256, 0, stream>>>(Wo, wob, (size_t)HID * (NH * HD));

  // 2) fused QKV projections (WMMA + async-LDS staging + RoPE/transpose epilogues)
  gemm_fused<0><<<dim3((NH * HD) / 64, Mrows / 256), 256, 0, stream>>>(
      xb, wqb, pos, qb, Mrows, NH * HD, HID, NH, 0.125f);
  gemm_fused<0><<<dim3((NKV * HD) / 64, Mrows / 256), 256, 0, stream>>>(
      xb, wkb, pos, kb, Mrows, NKV * HD, HID, NKV, 1.0f);
  gemm_fused<2><<<dim3((NKV * HD) / 64, Mrows / 256), 256, 0, stream>>>(
      xb, wvb, pos, vt, Mrows, NKV * HD, HID, NKV, 1.0f);

  // 3) attention (WMMA QK^T + PV, online softmax, WMMA row-sums, permlane max)
  attn_wmma<<<dim3(Tt / 16, NH, Bz), 32, 0, stream>>>(qb, kb, vt, ab);

  // 4) output projection, f32 straight to d_out
  gemm_fused<3><<<dim3(HID / 64, Mrows / 256), 256, 0, stream>>>(
      ab, wob, nullptr, out, Mrows, HID, NH * HD, 0, 1.0f);
}
